// FPU_4277787427678
// MI455X (gfx1250) — compile-verified, matrix operand order checked by reading, not run on hardware
//
#include <hip/hip_runtime.h>
#include <hip/hip_bf16.h>
#include <math.h>

// MI455X / gfx1250, wave32, WMMA f16 -> f32, async-to-LDS staging.
// Pipeline rewritten as GEMMs:
//   stage1+2 fused:  Y_pair[h,:] = sum_r X[h+r-48,:] @ A[pair,r]   (A = C_s @ G_r, precomputed f16, transposed)
//   stage3:          out[b,o,hw] = W[96x64] @ cat[b,hw,64]
// Workspace layout (bytes):   needs ~25.1 MB
//   [0,768)              circulant coeffs c[2][96] f32
//   [1024, 151568)       gabor g[4][97][97] f32
//   [152064, +14303232)  At f16 [8 pairs][97 r][96 n][96 k]
//   [+, +1179648)        x16 f16 [64 img][96][96]
//   [+, +9437184)        cat f16 [8 b][96 h][96 w][64 k]

typedef __attribute__((ext_vector_type(16))) _Float16 v16h;
typedef __attribute__((ext_vector_type(8)))  _Float16 v8h;
typedef __attribute__((ext_vector_type(8)))  float    v8f;

union Frag { v16h v; v8h h[2]; };

#define PI_F 3.14159265358979323846f

// ---- CDNA5 async-to-LDS path (guarded; falls back to VGPR-bounce copies) ----
#if defined(__has_builtin)
#if __has_builtin(__builtin_amdgcn_global_load_async_to_lds_b128) && \
    __has_builtin(__builtin_amdgcn_s_wait_asynccnt)
#define USE_ASYNC 1
#endif
#endif

#define AS1 __attribute__((address_space(1)))
#define AS3 __attribute__((address_space(3)))

// b128 payload type exactly as the builtin expects (GCC vector_size style)
typedef int b128_t __attribute__((vector_size(4 * sizeof(int))));

static __device__ __forceinline__ AS1 b128_t* to_global(const void* p) {
  return (AS1 b128_t*)(unsigned long long)(size_t)p;      // full 64-bit VA
}
static __device__ __forceinline__ AS3 b128_t* to_lds(void* p) {
  return (AS3 b128_t*)(unsigned)(size_t)p;                // flat addr low 32b = LDS offset
}

// copy one 9216-half (18432 B) A slice: threads 0..127 stage 9 x b128 chunks
static __device__ __forceinline__ void stage_slice(const _Float16* __restrict__ gsrc,
                                                   _Float16* __restrict__ ldst, int t) {
  if (t < 128) {
    const v8h* s = (const v8h*)gsrc;
    v8h* d = (v8h*)ldst;
    #pragma unroll
    for (int i = 0; i < 9; ++i) {
#ifdef USE_ASYNC
      __builtin_amdgcn_global_load_async_to_lds_b128(to_global(s + t + i * 128),
                                                     to_lds(d + t + i * 128), 0, 0);
#else
      d[t + i * 128] = s[t + i * 128];
#endif
    }
  }
}

static __device__ __forceinline__ void async_fence_lds() {
#ifdef USE_ASYNC
  __builtin_amdgcn_s_wait_asynccnt(0);   // all this wave's async LDS writes done
#endif
  __syncthreads();                       // visible to all waves
}

// ---------------- setup kernels (tiny, f32 VALU) ----------------

// c_s[d] = (1/96) * sum_f cos(2*pi*f*d/96 - pi*0.8*f^2/s^2)
__global__ void k_circ(float* __restrict__ c) {
  int t = threadIdx.x;           // 0..191
  int s = t / 96;                // scale-1
  int d = t % 96;
  float inv_s2 = 1.0f / (float)((s + 1) * (s + 1));
  float acc = 0.f;
  for (int f = 0; f < 96; ++f) {
    float ff = (float)f;
    acc += cosf(2.f * PI_F * ff * (float)d / 96.f - PI_F * 0.8f * ff * ff * inv_s2);
  }
  c[t] = acc * (1.f / 96.f);
}

// g[o][i][j], K=97, sigma=4, lambda=10, gamma=0.5, psi=0
__global__ void k_gabor(float* __restrict__ g) {
  int id = blockIdx.x * 256 + threadIdx.x;
  if (id >= 4 * 97 * 97) return;
  int o = id / (97 * 97);
  int rem = id % (97 * 97);
  int i = rem / 97, j = rem % 97;
  float theta = (float)o * (PI_F / 4.f);
  float y = (float)(i - 48), x = (float)(j - 48);
  float ct = cosf(theta), st = sinf(theta);
  float xr =  x * ct + y * st;
  float yr = -x * st + y * ct;
  g[id] = expf(-(xr * xr + 0.25f * yr * yr) * (1.f / 32.f)) * cosf(2.f * PI_F * xr * 0.1f);
}

// At[p][r][n][k] = sum_{w'} c_s[(w'-k) mod 96] * g[o][r][w'-n+48]
__global__ void k_buildA(const float* __restrict__ c, const float* __restrict__ g,
                         _Float16* __restrict__ At) {
  int p = blockIdx.x / 97;
  int r = blockIdx.x % 97;
  int s = p >> 2;
  int o = p & 3;
  const float* cs = c + s * 96;
  const float* gr = g + (o * 97 + r) * 97;
  _Float16* dst = At + (size_t)(p * 97 + r) * 9216;
  for (int e = 0; e < 36; ++e) {
    int idx = threadIdx.x + e * 256;   // < 9216
    int n = idx / 96, k = idx % 96;
    float acc = 0.f;
    for (int wp = 0; wp < 96; ++wp) {
      int gi = wp - n + 48;
      if (gi >= 0 && gi <= 96) {
        int ci = wp - k;
        ci += (ci >> 31) & 96;         // mod 96 for (-96,96)
        acc += cs[ci] * gr[gi];
      }
    }
    dst[idx] = (_Float16)acc;
  }
}

__global__ void k_cvt(const float* __restrict__ x, _Float16* __restrict__ x16) {
  int id = blockIdx.x * 256 + threadIdx.x;   // 589824 = 2304*256
  x16[id] = (_Float16)x[id];
}

// ---------------- main fused FrFT+Gabor GEMM ----------------
// grid (6 htiles, 16 groups of 4 images, 8 pairs), 256 thr = 8 waves.
// LDS: X 4*96*96 halves (73728B) + A double buffer 2*9216 halves (36864B)
__global__ void __launch_bounds__(256)
k_main(const _Float16* __restrict__ x16, const _Float16* __restrict__ At,
       _Float16* __restrict__ cat) {
  extern __shared__ _Float16 smem[];
  _Float16* Xl = smem;            // 36864 halves
  _Float16* Al = smem + 36864;    // 2 * 9216 halves

  const int t     = threadIdx.x;
  const int h0    = blockIdx.x * 16;
  const int grp   = blockIdx.y;
  const int pair  = blockIdx.z;

  // load 4 full images into LDS (4608 b128 chunks, async when available)
  {
    const v8h* src = (const v8h*)(x16 + (size_t)grp * 4 * 9216);
    v8h* dst = (v8h*)Xl;
    #pragma unroll
    for (int i = 0; i < 18; ++i) {
#ifdef USE_ASYNC
      __builtin_amdgcn_global_load_async_to_lds_b128(to_global(src + t + i * 256),
                                                     to_lds(dst + t + i * 256), 0, 0);
#else
      dst[t + i * 256] = src[t + i * 256];
#endif
    }
  }
  const _Float16* Abase = At + (size_t)pair * 97 * 9216;
  stage_slice(Abase, Al, t);      // preload A slice r=0
  async_fence_lds();

  const int wave  = t >> 5;
  const int lane  = t & 31;
  const int img   = wave >> 1;          // 0..3
  const int nbase = (wave & 1) * 48;    // N split across wave pairs
  const int mrow  = lane & 15;
  const int ksel  = lane >> 4;
  const int koff  = ksel * 8;

  v8f acc[3] = {};

  for (int r = 0; r < 97; ++r) {
    const _Float16* Acur = Al + (r & 1) * 9216;
    if (r + 1 < 97) {   // stage next A slice into the other buffer (ASYNCcnt path)
      stage_slice(Abase + (size_t)(r + 1) * 9216, Al + ((r + 1) & 1) * 9216, t);
      if (r + 2 < 97)   // global_prefetch_b8 the slice after that
        __builtin_prefetch(Abase + (size_t)(r + 2) * 9216 + t * 36, 0, 1);
    }

    const int row = h0 + mrow + r - 48;        // input row for this output row
    const bool valid = (unsigned)row < 96u;
    const _Float16* xrow = Xl + ((size_t)img * 96 + (valid ? row : 0)) * 96;
    const v8h zero = {};

    #pragma unroll
    for (int ks = 0; ks < 3; ++ks) {
      const int kb = ks * 32;
      Frag a;
      a.h[0] = *(const v8h*)(xrow + kb + koff);
      a.h[1] = *(const v8h*)(xrow + kb + koff + 16);
      if (!valid) { a.h[0] = zero; a.h[1] = zero; }   // zero pad rows, EXEC stays all-1s
      #pragma unroll
      for (int nt = 0; nt < 3; ++nt) {
        const int col = nbase + nt * 16 + mrow;
        const _Float16* bcol = Acur + col * 96 + kb + koff;
        Frag b;
        b.h[0] = *(const v8h*)(bcol);
        b.h[1] = *(const v8h*)(bcol + 16);
        acc[nt] = __builtin_amdgcn_wmma_f32_16x16x32_f16(
            false, a.v, false, b.v, (short)0, acc[nt], false, false);
      }
    }
    async_fence_lds();
  }

  // write to cat[b][h][w][k] (k = pair*8 + c contiguous for stage-3 fragments)
  const int id  = grp * 4 + img;
  const int b   = id >> 3;
  const int kch = pair * 8 + (id & 7);
  #pragma unroll
  for (int nt = 0; nt < 3; ++nt) {
    const int n = nbase + nt * 16 + mrow;
    #pragma unroll
    for (int i2 = 0; i2 < 8; ++i2) {
      const int hh = h0 + i2 + ksel * 8;    // D layout: VGPR i -> M=i (+8 for upper lanes)
      cat[(((size_t)b * 96 + hh) * 96 + n) * 64 + kch] = (_Float16)acc[nt][i2];
    }
  }
}

// ---------------- 1x1 mixing GEMM: out[b,o,hw] = W[96x64] @ cat[b,hw,:] ----------------
__global__ void __launch_bounds__(256)
k_final(const float* __restrict__ wpwc, const _Float16* __restrict__ cat,
        float* __restrict__ out) {
  __shared__ _Float16 Wl[96 * 64];
  const int t = threadIdx.x;
  #pragma unroll
  for (int i = 0; i < 24; ++i) Wl[t + i * 256] = (_Float16)wpwc[t + i * 256];
  __syncthreads();

  const int b    = blockIdx.y;
  const int wave = t >> 5, lane = t & 31;
  const int n0   = blockIdx.x * 128 + wave * 16;   // 72 blocks * 128 = 9216
  const int mrow = lane & 15, ksel = lane >> 4, koff = ksel * 8;

  v8f acc[6] = {};
  #pragma unroll
  for (int kb = 0; kb < 64; kb += 32) {
    Frag bf;
    const _Float16* bp = cat + ((size_t)b * 9216 + n0 + mrow) * 64 + kb + koff;
    bf.h[0] = *(const v8h*)bp;
    bf.h[1] = *(const v8h*)(bp + 16);
    #pragma unroll
    for (int mt = 0; mt < 6; ++mt) {
      Frag af;
      const _Float16* ap = Wl + (mt * 16 + mrow) * 64 + kb + koff;
      af.h[0] = *(const v8h*)ap;
      af.h[1] = *(const v8h*)(ap + 16);
      acc[mt] = __builtin_amdgcn_wmma_f32_16x16x32_f16(
          false, af.v, false, bf.v, (short)0, acc[mt], false, false);
    }
  }
  #pragma unroll
  for (int mt = 0; mt < 6; ++mt) {
    #pragma unroll
    for (int i2 = 0; i2 < 8; ++i2) {
      const int o = mt * 16 + i2 + ksel * 8;
      out[((size_t)b * 96 + o) * 9216 + n0 + mrow] = acc[mt][i2];
    }
  }
}

// ---------------- host ----------------
extern "C" void kernel_launch(void* const* d_in, const int* in_sizes, int n_in,
                              void* d_out, int out_size, void* d_ws, size_t ws_size,
                              hipStream_t stream) {
  (void)in_sizes; (void)n_in; (void)out_size; (void)ws_size;
  const float* x    = (const float*)d_in[0];   // [8,8,96,96] f32
  const float* wpwc = (const float*)d_in[1];   // [96,64] f32
  float* out = (float*)d_out;                  // [8,96,96,96] f32

  char* ws = (char*)d_ws;
  float*     c_s = (float*)(ws + 0);                      // 768 B
  float*     gab = (float*)(ws + 1024);                   // 150544 B
  _Float16*  At  = (_Float16*)(ws + 152064);              // 14303232 B
  _Float16*  x16 = (_Float16*)(ws + 152064 + 14303232);   // 1179648 B
  _Float16*  cat = (_Float16*)(ws + 152064 + 14303232 + 1179648); // 9437184 B

  hipLaunchKernelGGL(k_circ,   dim3(1),        dim3(192), 0, stream, c_s);
  hipLaunchKernelGGL(k_gabor,  dim3(148),      dim3(256), 0, stream, gab);
  hipLaunchKernelGGL(k_buildA, dim3(776),      dim3(256), 0, stream, c_s, gab, At);
  hipLaunchKernelGGL(k_cvt,    dim3(2304),     dim3(256), 0, stream, x, x16);
  hipLaunchKernelGGL(k_main,   dim3(6, 16, 8), dim3(256),
                     (size_t)(36864 + 2 * 9216) * sizeof(_Float16), stream,
                     x16, At, cat);
  hipLaunchKernelGGL(k_final,  dim3(72, 8),    dim3(256), 0, stream, wpwc, cat, out);
}
